// Block_mem_43336220016755
// MI455X (gfx1250) — matrix operand order repaired and checked by reading, not compiled, not run on hardware
//
#include <hip/hip_runtime.h>

// ---------------------------------------------------------------------------
// Cosine top-10 retrieval on gfx1250 (MI455X, wave32, WMMA bf16).
//   B=4096 queries, DIM=512, K=65536 bank rows, TOP_N=10.
// Pipeline:
//   1) normalize rows of query+bank, convert to bf16 (RNE) into workspace
//   2) tiled WMMA bf16 GEMM (f32 accum) + per-1024-key-chunk top-10 select
//   3) per-query reduction of 640 candidates -> top-10, gather raw rows
// Workspace layout (bytes):
//   qn_bf16 : 4096*512*2           =  4,194,304
//   mn_bf16 : 65536*512*2          = 67,108,864
//   candVal : 4096*64*10*4         = 10,485,760
//   candIdx : 4096*64*10*4         = 10,485,760   (total ~92.3 MB)
// ---------------------------------------------------------------------------

#define NB      4096
#define NK      65536
#define DIMS    512
#define TOPN    10
#define CHUNK   1024                 // keys per block in pass 2
#define NCHUNK  (NK / CHUNK)         // 64
#define QTILE   16

typedef __bf16 v16bf __attribute__((ext_vector_type(16)));
typedef float  v8f   __attribute__((ext_vector_type(8)));

union FragAB { v16bf v; uint4 q[2]; };

__device__ __forceinline__ unsigned short f32_to_bf16_rne(float f) {
    unsigned int u = __float_as_uint(f);
    u += 0x7FFFu + ((u >> 16) & 1u);       // round to nearest even
    return (unsigned short)(u >> 16);
}

// ---------------------------------------------------------------------------
// Pass 1: L2-normalize each row, convert to bf16. One wave per row.
// Rows [0,4096) -> query, rows [4096, 69632) -> bank.
// ---------------------------------------------------------------------------
__global__ __launch_bounds__(256)
void normalize_bf16_kernel(const float* __restrict__ query,
                           const float* __restrict__ queue,
                           unsigned short* __restrict__ qn,
                           unsigned short* __restrict__ mn)
{
    const int wave = threadIdx.x >> 5;
    const int lane = threadIdx.x & 31;
    const int row  = blockIdx.x * 8 + wave;          // 8 waves per block

    const float* src;
    unsigned short* dst;
    if (row < NB) { src = query + (size_t)row * DIMS;        dst = qn + (size_t)row * DIMS; }
    else          { src = queue + (size_t)(row - NB) * DIMS; dst = mn + (size_t)(row - NB) * DIMS; }

    const float4* s4 = (const float4*)src;
    float4 x[4];
    float  ss = 0.f;
    #pragma unroll
    for (int j = 0; j < 4; ++j) {                    // 512 f32 = 128 float4 = 32 lanes * 4
        x[j] = s4[lane + j * 32];
        ss += x[j].x * x[j].x + x[j].y * x[j].y + x[j].z * x[j].z + x[j].w * x[j].w;
    }
    #pragma unroll
    for (int off = 16; off > 0; off >>= 1) ss += __shfl_xor(ss, off, 32);

    const float inv = 1.0f / fmaxf(sqrtf(ss), 1e-12f);

    uint2* d2 = (uint2*)dst;                          // 4 bf16 per uint2
    #pragma unroll
    for (int j = 0; j < 4; ++j) {
        uint2 p;
        p.x = (unsigned)f32_to_bf16_rne(x[j].x * inv) |
              ((unsigned)f32_to_bf16_rne(x[j].y * inv) << 16);
        p.y = (unsigned)f32_to_bf16_rne(x[j].z * inv) |
              ((unsigned)f32_to_bf16_rne(x[j].w * inv) << 16);
        d2[lane + j * 32] = p;
    }
}

// ---------------------------------------------------------------------------
// Pass 2: per (16-query tile, 1024-key chunk): bf16 WMMA GEMM -> LDS scores,
// then per-row top-10 via iterative wave-max. Block = 256 thr = 8 waves.
// Wave w computes keys [w*128, (w+1)*128) of the chunk as 8 WMMA tiles.
// Inner loop: all 16 B-fragment b128 loads are clustered ahead of the
// 8-WMMA chain so the backend can use partial s_wait_loadcnt and overlap
// remaining loads with earlier WMMAs.  k-loop pinned to unroll(1) to keep
// live fragments (~210 VGPRs) inside the direct-addressable window.
// ---------------------------------------------------------------------------
__global__ __launch_bounds__(256)
void cosine_topk_chunk_kernel(const unsigned short* __restrict__ qn,
                              const unsigned short* __restrict__ mn,
                              float* __restrict__ candVal,
                              int*   __restrict__ candIdx)
{
    __shared__ float s_scores[QTILE * CHUNK];        // 64 KB

    const int tid   = threadIdx.x;
    const int wave  = tid >> 5;
    const int lane  = tid & 31;
    const int chunk = blockIdx.x;                    // 0..63
    const int qtile = blockIdx.y;                    // 0..255

    // A fragment (16x32 bf16): lane holds row (lane&15),
    // K = (lane>>4)*8 + {0..7} and + {16..23}  (two contiguous 16B loads)
    const int am = lane & 15;
    const int ak = (lane >> 4) * 8;
    const unsigned short* arow = qn + (size_t)(qtile * QTILE + am) * DIMS;

    // B fragment (32x16 bf16): lane holds col (lane&15) = bank row,
    // K = (lane>>4)*8 + {0..7} and + {16..23}  (two contiguous 16B loads)
    const int bn = lane & 15;
    const int bk = (lane >> 4) * 8;
    const unsigned short* brow0 =
        mn + (size_t)(chunk * CHUNK + wave * 128 + bn) * DIMS;

    v8f acc[8];
    #pragma unroll
    for (int t = 0; t < 8; ++t) acc[t] = (v8f){0.f,0.f,0.f,0.f,0.f,0.f,0.f,0.f};

    #pragma unroll 1
    for (int k = 0; k < DIMS; k += 32) {
        FragAB a;
        a.q[0] = *(const uint4*)(arow + k + ak);
        a.q[1] = *(const uint4*)(arow + k + ak + 16);

        FragAB b[8];
        #pragma unroll
        for (int t = 0; t < 8; ++t) {
            const unsigned short* bp = brow0 + (size_t)t * 16 * DIMS + k + bk;
            b[t].q[0] = *(const uint4*)(bp);
            b[t].q[1] = *(const uint4*)(bp + 16);
        }

        #pragma unroll
        for (int t = 0; t < 8; ++t)
            acc[t] = __builtin_amdgcn_wmma_f32_16x16x32_bf16(
                         false, a.v, false, b[t].v, (short)0, acc[t], false, false);
    }

    // C/D layout: lane holds col n=(lane&15), rows m=(lane>>4)*8 + r (r=VGPR idx)
    #pragma unroll
    for (int t = 0; t < 8; ++t) {
        const int kk    = wave * 128 + t * 16 + (lane & 15);
        const int mbase = (lane >> 4) * 8;
        #pragma unroll
        for (int r = 0; r < 8; ++r)
            s_scores[(mbase + r) * CHUNK + kk] = acc[t][r];
    }
    __syncthreads();

    // Top-10 per query row over this chunk. Wave w handles rows 2w, 2w+1.
    for (int rr = 0; rr < 2; ++rr) {
        const int row = wave * 2 + rr;
        float v[32];
        #pragma unroll
        for (int j = 0; j < 32; ++j)
            v[j] = s_scores[row * CHUNK + lane + j * 32]; // key = lane + j*32

        unsigned removed = 0;
        const int    q        = qtile * QTILE + row;
        const size_t candBase = ((size_t)q * NCHUNK + chunk) * TOPN;

        for (int sel = 0; sel < TOPN; ++sel) {
            float bv = -3.402823466e38f;
            int   bkk = 0x7fffffff;
            #pragma unroll
            for (int j = 0; j < 32; ++j) {
                if (!((removed >> j) & 1u)) {
                    const int key = lane + j * 32;
                    if (v[j] > bv || (v[j] == bv && key < bkk)) { bv = v[j]; bkk = key; }
                }
            }
            #pragma unroll
            for (int off = 16; off > 0; off >>= 1) {
                const float ov = __shfl_xor(bv, off, 32);
                const int   ok = __shfl_xor(bkk, off, 32);
                if (ov > bv || (ov == bv && ok < bkk)) { bv = ov; bkk = ok; }
            }
            if ((bkk & 31) == lane) removed |= 1u << (bkk >> 5);   // drop winner
            if (lane == 0) {
                candVal[candBase + sel] = bv;
                candIdx[candBase + sel] = chunk * CHUNK + bkk;
            }
        }
    }
}

// ---------------------------------------------------------------------------
// Pass 3: one block per query. Wave 0 reduces 640 candidates -> ordered
// top-10 (desc value, ties -> lower index), then all threads gather the raw
// bank rows to d_out [B][10][512].
// ---------------------------------------------------------------------------
__global__ __launch_bounds__(256)
void topk_reduce_gather_kernel(const float* __restrict__ candVal,
                               const int*   __restrict__ candIdx,
                               const float* __restrict__ queue,
                               float*       __restrict__ out)
{
    __shared__ int s_top[TOPN];
    const int q   = blockIdx.x;
    const int tid = threadIdx.x;

    if (tid < 32) {
        const int lane = tid;
        const size_t base = (size_t)q * (NCHUNK * TOPN);   // 640 candidates
        float v[20]; int gi[20];
        #pragma unroll
        for (int j = 0; j < 20; ++j) {
            v[j]  = candVal[base + lane + j * 32];
            gi[j] = candIdx[base + lane + j * 32];
        }
        unsigned removed = 0;
        for (int sel = 0; sel < TOPN; ++sel) {
            float bv = -3.402823466e38f;
            int bidx = 0x7fffffff, bp = -1;
            #pragma unroll
            for (int j = 0; j < 20; ++j) {
                if (!((removed >> j) & 1u)) {
                    if (v[j] > bv || (v[j] == bv && gi[j] < bidx)) {
                        bv = v[j]; bidx = gi[j]; bp = lane + j * 32;
                    }
                }
            }
            #pragma unroll
            for (int off = 16; off > 0; off >>= 1) {
                const float ov = __shfl_xor(bv, off, 32);
                const int   oi = __shfl_xor(bidx, off, 32);
                const int   op = __shfl_xor(bp, off, 32);
                if (ov > bv || (ov == bv && oi < bidx)) { bv = ov; bidx = oi; bp = op; }
            }
            if (bp >= 0 && (bp & 31) == lane) removed |= 1u << (bp >> 5);
            if (lane == 0) s_top[sel] = bidx;
        }
    }
    __syncthreads();

    // gather 10 rows x 512 f32 = 1280 float4
    const float4* src = (const float4*)queue;
    float4* dst = (float4*)(out + (size_t)q * TOPN * DIMS);
    for (int i = tid; i < TOPN * (DIMS / 4); i += 256) {
        const int nrow = i >> 7;          // / 128
        const int col4 = i & 127;
        dst[i] = src[(size_t)s_top[nrow] * (DIMS / 4) + col4];
    }
}

// ---------------------------------------------------------------------------
extern "C" void kernel_launch(void* const* d_in, const int* in_sizes, int n_in,
                              void* d_out, int out_size, void* d_ws, size_t ws_size,
                              hipStream_t stream)
{
    const float* query = (const float*)d_in[0];   // [4096, 512]
    const float* queue = (const float*)d_in[1];   // [65536, 512]

    unsigned short* qn = (unsigned short*)d_ws;
    unsigned short* mn = qn + (size_t)NB * DIMS;
    float* candVal = (float*)(mn + (size_t)NK * DIMS);
    int*   candIdx = (int*)(candVal + (size_t)NB * NCHUNK * TOPN);
    float* out = (float*)d_out;

    // Pass 1: 69632 rows, 8 rows (waves) per 256-thread block
    normalize_bf16_kernel<<<dim3((NB + NK) / 8), 256, 0, stream>>>(query, queue, qn, mn);

    // Pass 2: 64 key-chunks x 256 query-tiles
    cosine_topk_chunk_kernel<<<dim3(NCHUNK, NB / QTILE), 256, 0, stream>>>(
        qn, mn, candVal, candIdx);

    // Pass 3: one block per query
    topk_reduce_gather_kernel<<<dim3(NB), 256, 0, stream>>>(candVal, candIdx, queue, out);
}
